// AudioViTEncoder_85237920957035
// MI455X (gfx1250) — matrix operand order repaired and checked by reading, not compiled, not run on hardware
//
#include <hip/hip_runtime.h>
#include <hip/hip_bf16.h>
#include <math.h>

// ---------------------------------------------------------------------------
// AudioViT encoder forward for MI455X (gfx1250, wave32, WMMA).
// All GEMMs + attention einsums use v_wmma_f32_16x16x32_bf16.
// RoPE + head-reshape fused into the Q/K/V projection GEMM epilogues.
// ---------------------------------------------------------------------------

#define BB   32
#define SS   512
#define DDIM 256
#define HH   16
#define HDIM 16
#define LLAY 8
#define FFD  512
#define LATD 256
#define PDIM 64
#define ROWS (BB * SS)          // 16384 token rows

typedef __attribute__((ext_vector_type(16))) __bf16 v16bf;
typedef __attribute__((ext_vector_type(8)))  float  v8f;

union BF16x16 { v16bf v; unsigned short u[16]; uint4 q2[2]; };

__device__ __forceinline__ unsigned short f2bf(float f) {
    union { float f; unsigned u; } x; x.f = f;
    unsigned r = x.u + 0x7FFFu + ((x.u >> 16) & 1u);   // round-to-nearest-even
    return (unsigned short)(r >> 16);
}
__device__ __forceinline__ unsigned pack2(float lo, float hi) {
    return (unsigned)f2bf(lo) | ((unsigned)f2bf(hi) << 16);
}
__device__ __forceinline__ float gelu_f(float x) {     // tanh approximation (jax default)
    float x3 = x * x * x;
    return 0.5f * x * (1.f + tanhf(0.7978845608028654f * (x + 0.044715f * x3)));
}

// ---------------------------------------------------------------------------
// Epilogue store for one 16-col tile of a wave's C fragment.
// OUTMODE 0: plain C[M,N] (+bias, +resid, +gelu per flags)
// OUTMODE 1: +bias, RoPE, scatter to [B,H,S,HD]   (16-col tile == one head)
// OUTMODE 2: +bias, scatter to [B,H,S,HD] (no rope)
// ---------------------------------------------------------------------------
template<bool BIAS, bool GELU, bool RESID, int OUTMODE>
__device__ __forceinline__ void gemm_store(v8f acc, int mb, int ngl, int r,
                                           const float* __restrict__ bias,
                                           const float* __restrict__ resid,
                                           float* __restrict__ C, int N)
{
    const float bv = BIAS ? bias[ngl] : 0.f;
    if (OUTMODE == 0) {
        #pragma unroll
        for (int i = 0; i < 8; ++i) {
            float xv = acc[i] + bv;
            if (RESID) xv += resid[(size_t)(mb + i) * N + ngl];
            if (GELU)  xv = gelu_f(xv);
            C[(size_t)(mb + i) * N + ngl] = xv;
        }
    } else {
        const int h = ngl >> 4;                 // head index
        const int d = r;                        // dim within head (== ngl & 15)
        #pragma unroll
        for (int i = 0; i < 8; ++i) {
            int row = mb + i;                   // b*S + s
            int s   = row & 511;
            int bbb = row >> 9;
            float xv = acc[i] + bv;
            float ov = xv;
            if (OUTMODE == 1) {                 // rope: pair (d, d^8) via lane shuffle
                float p  = __shfl_xor(xv, 8, 32);
                int   j  = d & 7;
                float freq = __expf(-1.1512925464970231f * (float)j);  // 10000^(-j/8)
                float ang  = (float)s * freq;
                float sn, cs; __sincosf(ang, &sn, &cs);
                ov = (d < 8) ? (xv * cs - p * sn) : (xv * cs + p * sn);
            }
            C[(((size_t)(bbb * 16 + h) * 512) + s) * 16 + d] = ov;
        }
    }
}

// ---------------------------------------------------------------------------
// Main bf16-WMMA GEMM: 128 threads / 4 waves (2x2), 64x64 block tile, each
// wave owns a 32x32 panel -> 4 WMMAs per K-step, every fragment reused 2x.
// Double-buffered LDS, one barrier per K-step, prefetch of the K+2 tile.
// M,N % 64 == 0, K % 32 == 0.
// ---------------------------------------------------------------------------
template<bool BIAS, bool GELU, bool RESID, int OUTMODE>
__global__ __launch_bounds__(128)
void gemm64_bf16_wmma(const float* __restrict__ A, const float* __restrict__ Bw,
                      const float* __restrict__ bias, const float* __restrict__ resid,
                      float* __restrict__ C, int M, int N, int K)
{
    __shared__ unsigned short As[2][64][40];   // [stage][m][k] bf16 (80B rows, 16B aligned)
    __shared__ unsigned short Bs[2][64][40];   // [stage][n][k] bf16 (transposed on load)

    const int tid  = threadIdx.x;
    const int wave = tid >> 5;
    const int lane = tid & 31;
    const int wm   = (wave >> 1) << 5;      // 0,32
    const int wn   = (wave & 1) << 5;       // 0,32
    const int m0   = blockIdx.y << 6;
    const int n0   = blockIdx.x << 6;
    const int hlf  = lane >> 4;
    const int r    = lane & 15;

    const int arow = tid >> 1;              // 0..63
    const int acol = (tid & 1) << 4;        // 0,16
    const int brow = tid >> 2;              // 0..31 (k)
    const int bcol = (tid & 3) << 4;        // 0,16,32,48 (n)

    const float* aptr = A + (size_t)(m0 + arow) * K + acol;
    const float* bptr = Bw + (size_t)brow * N + n0 + bcol;

    // ---- preload K-step 0 into stage 0 ----
    {
        float4 a0 = *(const float4*)(aptr);
        float4 a1 = *(const float4*)(aptr + 4);
        float4 a2 = *(const float4*)(aptr + 8);
        float4 a3 = *(const float4*)(aptr + 12);
        uint4 p0, p1;
        p0.x = pack2(a0.x, a0.y); p0.y = pack2(a0.z, a0.w);
        p0.z = pack2(a1.x, a1.y); p0.w = pack2(a1.z, a1.w);
        p1.x = pack2(a2.x, a2.y); p1.y = pack2(a2.z, a2.w);
        p1.z = pack2(a3.x, a3.y); p1.w = pack2(a3.z, a3.w);
        *(uint4*)&As[0][arow][acol]     = p0;
        *(uint4*)&As[0][arow][acol + 8] = p1;

        float4 b0 = *(const float4*)(bptr);
        float4 b1 = *(const float4*)(bptr + 4);
        float4 b2 = *(const float4*)(bptr + 8);
        float4 b3 = *(const float4*)(bptr + 12);
        float bvv[16] = {b0.x, b0.y, b0.z, b0.w, b1.x, b1.y, b1.z, b1.w,
                         b2.x, b2.y, b2.z, b2.w, b3.x, b3.y, b3.z, b3.w};
        #pragma unroll
        for (int j = 0; j < 16; ++j) Bs[0][bcol + j][brow] = f2bf(bvv[j]);
    }
    __syncthreads();

    v8f c00 = {}, c01 = {}, c10 = {}, c11 = {};
    const int nsteps = K >> 5;

    for (int s = 0; s < nsteps; ++s) {
        const int cur = s & 1;

        // issue next tile's global loads early (overlap with WMMAs below)
        float4 a0, a1, a2, a3, b0, b1, b2, b3;
        const bool have_next = (s + 1 < nsteps);
        if (have_next) {
            const float* ap = aptr + (size_t)(s + 1) * 32;
            const float* bp = bptr + (size_t)(s + 1) * 32 * N;
            a0 = *(const float4*)(ap);      a1 = *(const float4*)(ap + 4);
            a2 = *(const float4*)(ap + 8);  a3 = *(const float4*)(ap + 12);
            b0 = *(const float4*)(bp);      b1 = *(const float4*)(bp + 4);
            b2 = *(const float4*)(bp + 8);  b3 = *(const float4*)(bp + 12);
            if (s + 2 < nsteps) {                     // K+2 tile -> global_prefetch_b8
                __builtin_prefetch((const void*)(aptr + (size_t)(s + 2) * 32), 0, 1);
                __builtin_prefetch((const void*)(bptr + (size_t)(s + 2) * 32 * N), 0, 1);
            }
        }

        // ---- fragments (16B LDS vector reads) ----
        BF16x16 af0, af1, bf0, bf1;
        af0.q2[0] = *(const uint4*)&As[cur][wm + r][hlf << 3];
        af0.q2[1] = *(const uint4*)&As[cur][wm + r][(hlf << 3) + 16];
        af1.q2[0] = *(const uint4*)&As[cur][wm + 16 + r][hlf << 3];
        af1.q2[1] = *(const uint4*)&As[cur][wm + 16 + r][(hlf << 3) + 16];
        bf0.q2[0] = *(const uint4*)&Bs[cur][wn + r][hlf << 3];
        bf0.q2[1] = *(const uint4*)&Bs[cur][wn + r][(hlf << 3) + 16];
        bf1.q2[0] = *(const uint4*)&Bs[cur][wn + 16 + r][hlf << 3];
        bf1.q2[1] = *(const uint4*)&Bs[cur][wn + 16 + r][(hlf << 3) + 16];

        c00 = __builtin_amdgcn_wmma_f32_16x16x32_bf16(false, af0.v, false, bf0.v,
                                                      (short)0, c00, false, false);
        c01 = __builtin_amdgcn_wmma_f32_16x16x32_bf16(false, af0.v, false, bf1.v,
                                                      (short)0, c01, false, false);
        c10 = __builtin_amdgcn_wmma_f32_16x16x32_bf16(false, af1.v, false, bf0.v,
                                                      (short)0, c10, false, false);
        c11 = __builtin_amdgcn_wmma_f32_16x16x32_bf16(false, af1.v, false, bf1.v,
                                                      (short)0, c11, false, false);

        if (have_next) {
            const int nxt = cur ^ 1;
            uint4 p0, p1;
            p0.x = pack2(a0.x, a0.y); p0.y = pack2(a0.z, a0.w);
            p0.z = pack2(a1.x, a1.y); p0.w = pack2(a1.z, a1.w);
            p1.x = pack2(a2.x, a2.y); p1.y = pack2(a2.z, a2.w);
            p1.z = pack2(a3.x, a3.y); p1.w = pack2(a3.z, a3.w);
            *(uint4*)&As[nxt][arow][acol]     = p0;
            *(uint4*)&As[nxt][arow][acol + 8] = p1;
            float bvv[16] = {b0.x, b0.y, b0.z, b0.w, b1.x, b1.y, b1.z, b1.w,
                             b2.x, b2.y, b2.z, b2.w, b3.x, b3.y, b3.z, b3.w};
            #pragma unroll
            for (int j = 0; j < 16; ++j) Bs[nxt][bcol + j][brow] = f2bf(bvv[j]);
        }
        __syncthreads();
    }

    // ---- epilogue: four 16x16 tiles per wave ----
    const int mb0 = m0 + wm + (hlf << 3);
    const int mb1 = mb0 + 16;
    const int ng0 = n0 + wn + r;
    const int ng1 = ng0 + 16;
    gemm_store<BIAS, GELU, RESID, OUTMODE>(c00, mb0, ng0, r, bias, resid, C, N);
    gemm_store<BIAS, GELU, RESID, OUTMODE>(c01, mb0, ng1, r, bias, resid, C, N);
    gemm_store<BIAS, GELU, RESID, OUTMODE>(c10, mb1, ng0, r, bias, resid, C, N);
    gemm_store<BIAS, GELU, RESID, OUTMODE>(c11, mb1, ng1, r, bias, resid, C, N);
}

// ---------------------------------------------------------------------------
// Small-M bf16-WMMA GEMM (32x32 block tile, 128 threads) for M=32 cases:
// latent projection (split-K partials) and final head.
// ---------------------------------------------------------------------------
template<bool BIAS, bool SPLITK>
__global__ __launch_bounds__(128)
void gemm32_bf16_wmma(const float* __restrict__ A, const float* __restrict__ Bw,
                      const float* __restrict__ bias, float* __restrict__ C,
                      int M, int N, int K, int kchunk)
{
    __shared__ unsigned short As[32][40];
    __shared__ unsigned short Bs[32][40];

    const int tid  = threadIdx.x;
    const int wave = tid >> 5;
    const int lane = tid & 31;
    const int wm   = (wave >> 1) << 4;
    const int wn   = (wave & 1) << 4;
    const int m0   = blockIdx.y << 5;
    const int n0   = blockIdx.x << 5;
    const int hlf  = lane >> 4;
    const int r    = lane & 15;

    int k0 = 0, k1 = K;
    if (SPLITK) { k0 = blockIdx.z * kchunk; k1 = k0 + kchunk; if (k1 > K) k1 = K; }

    const int lrow = tid >> 2;
    const int lcol = (tid & 3) << 3;

    v8f acc = {};

    for (int kk = k0; kk < k1; kk += 32) {
        {
            const float* ap = A + (size_t)(m0 + lrow) * K + kk + lcol;
            float4 a0 = *(const float4*)(ap);
            float4 a1 = *(const float4*)(ap + 4);
            uint4 pk;
            pk.x = pack2(a0.x, a0.y); pk.y = pack2(a0.z, a0.w);
            pk.z = pack2(a1.x, a1.y); pk.w = pack2(a1.z, a1.w);
            *(uint4*)&As[lrow][lcol] = pk;
        }
        {
            const float* bp = Bw + (size_t)(kk + lrow) * N + n0 + lcol;
            float4 b0 = *(const float4*)(bp);
            float4 b1 = *(const float4*)(bp + 4);
            float bvv[8] = {b0.x, b0.y, b0.z, b0.w, b1.x, b1.y, b1.z, b1.w};
            #pragma unroll
            for (int j = 0; j < 8; ++j) Bs[lcol + j][lrow] = f2bf(bvv[j]);
        }
        __syncthreads();

        BF16x16 af, bf;
        af.q2[0] = *(const uint4*)&As[wm + r][hlf << 3];
        af.q2[1] = *(const uint4*)&As[wm + r][(hlf << 3) + 16];
        bf.q2[0] = *(const uint4*)&Bs[wn + r][hlf << 3];
        bf.q2[1] = *(const uint4*)&Bs[wn + r][(hlf << 3) + 16];

        acc = __builtin_amdgcn_wmma_f32_16x16x32_bf16(false, af.v, false, bf.v,
                                                      (short)0, acc, false, false);
        __syncthreads();
    }

    const int n  = n0 + wn + r;
    const int mb = m0 + wm + (hlf << 3);
    if (SPLITK) {
        float* Cp = C + (size_t)blockIdx.z * M * N;
        #pragma unroll
        for (int i = 0; i < 8; ++i)
            Cp[(size_t)(mb + i) * N + n] = acc[i];
    } else {
        const float bv = BIAS ? bias[n] : 0.f;
        #pragma unroll
        for (int i = 0; i < 8; ++i)
            C[(size_t)(mb + i) * N + n] = acc[i] + bv;
    }
}

// ---------------------------------------------------------------------------
// Patch extraction + LayerNorm(PD=64). One wave per (b,s) row, 2 elems/lane.
// ---------------------------------------------------------------------------
__global__ __launch_bounds__(256)
void patch_ln_kernel(const float* __restrict__ x, const float* __restrict__ g,
                     const float* __restrict__ be, float* __restrict__ out)
{
    const int wave = threadIdx.x >> 5, lane = threadIdx.x & 31;
    const int row  = blockIdx.x * 8 + wave;      // b*S + s
    const int bb   = row >> 9, s = row & 511;
    const int ph   = s >> 5,  pw = s & 31;
    float v[2];
    #pragma unroll
    for (int t = 0; t < 2; ++t) {
        int pd = lane + t * 32;
        int p1 = pd >> 3, p2 = pd & 7;
        v[t] = x[((size_t)bb * 128 + ph * 8 + p1) * 256 + pw * 8 + p2];
    }
    float s1 = v[0] + v[1], s2 = v[0] * v[0] + v[1] * v[1];
    #pragma unroll
    for (int m = 16; m; m >>= 1) { s1 += __shfl_xor(s1, m, 32); s2 += __shfl_xor(s2, m, 32); }
    float mean = s1 * (1.f / 64.f);
    float var  = s2 * (1.f / 64.f) - mean * mean;
    float inv  = rsqrtf(var + 1e-5f);
    #pragma unroll
    for (int t = 0; t < 2; ++t) {
        int pd = lane + t * 32;
        out[(size_t)row * 64 + pd] = (v[t] - mean) * inv * g[pd] + be[pd];
    }
}

// ---------------------------------------------------------------------------
// LayerNorm over last dim (wave per row).
// ---------------------------------------------------------------------------
__global__ __launch_bounds__(256)
void layernorm_kernel(const float* __restrict__ in, const float* __restrict__ g,
                      const float* __restrict__ be, float* __restrict__ out,
                      int rows, int dim)
{
    const int wave = threadIdx.x >> 5, lane = threadIdx.x & 31;
    const int row  = blockIdx.x * 8 + wave;
    if (row >= rows) return;
    const float* xp = in + (size_t)row * dim;
    float s1 = 0.f, s2 = 0.f;
    for (int i = lane; i < dim; i += 32) { float v = xp[i]; s1 += v; s2 += v * v; }
    #pragma unroll
    for (int m = 16; m; m >>= 1) { s1 += __shfl_xor(s1, m, 32); s2 += __shfl_xor(s2, m, 32); }
    float mean = s1 / (float)dim;
    float var  = s2 / (float)dim - mean * mean;
    float inv  = rsqrtf(var + 1e-5f);
    float* op = out + (size_t)row * dim;
    for (int i = lane; i < dim; i += 32) op[i] = (xp[i] - mean) * inv * g[i] + be[i];
}

// ---------------------------------------------------------------------------
// Flash attention with ALiBi. Block = 4 waves = 64 q rows of one (b,h);
// 16 iterations of 32 keys. Per iteration: 2 WMMAs for QK^T (K=HD=16 padded)
// + 1 full-K WMMA for P@V (P 16x32, V 32x16 - no zero padding).
// ---------------------------------------------------------------------------
__global__ __launch_bounds__(128)
void flash_attn_kernel(const float* __restrict__ q, const float* __restrict__ k,
                       const float* __restrict__ v, float* __restrict__ out)
{
    __shared__ unsigned short ks[32][24];      // [key][hd] bf16 (48B rows)
    __shared__ unsigned short vsT[16][40];     // [hd][key0..31] bf16 (80B rows)
    __shared__ unsigned short ps[4][16][40];   // per-wave probs [qrow][key0..31]

    const int tid  = threadIdx.x, wave = tid >> 5, lane = tid & 31;
    const int hlf  = lane >> 4, r = lane & 15;
    const int bh   = blockIdx.x >> 3, qt = blockIdx.x & 7;
    const int bb   = bh >> 4, h = bh & 15;
    const int q0   = qt * 64 + wave * 16;
    const float slope = exp2f(-0.5f * (float)(h + 1));   // 2^(-8*(h+1)/H)

    const float* qp = q + (size_t)bh * 512 * 16;
    const float* kp = k + (size_t)bh * 512 * 16;
    const float* vp = v + (size_t)bh * 512 * 16;

    // q fragment (A layout), 1/sqrt(HD)=0.25 folded; K 16..31 zero pad.
    BF16x16 qa;
    #pragma unroll
    for (int j = 0; j < 8; ++j) {
        qa.u[j]     = f2bf(0.25f * qp[(size_t)(q0 + r) * 16 + hlf * 8 + j]);
        qa.u[8 + j] = 0;
    }

    float rm[8], rl[8];
    v8f oacc = {};
    #pragma unroll
    for (int i = 0; i < 8; ++i) { rm[i] = -1e30f; rl[i] = 0.f; }

    for (int kt = 0; kt < 16; ++kt) {
        __syncthreads();
        for (int e = tid; e < 512; e += 128) {           // stage 32 keys of k,v
            int key = e >> 4, hd = e & 15;
            float kvv = kp[(size_t)(kt * 32 + key) * 16 + hd];
            float vvv = vp[(size_t)(kt * 32 + key) * 16 + hd];
            ks[key][hd]  = f2bf(kvv);
            vsT[hd][key] = f2bf(vvv);
        }
        if (kt + 1 < 16) {                               // prefetch next key block
            __builtin_prefetch((const void*)(kp + (size_t)(kt + 1) * 32 * 16 + tid * 4), 0, 1);
            __builtin_prefetch((const void*)(vp + (size_t)(kt + 1) * 32 * 16 + tid * 4), 0, 1);
        }
        __syncthreads();

        // QK^T for two key tiles (B: col n = key, K = hd 0..15 real, 16..31 zero)
        BF16x16 kf0, kf1;
        kf0.q2[0] = *(const uint4*)&ks[r][hlf << 3];
        kf1.q2[0] = *(const uint4*)&ks[16 + r][hlf << 3];
        {
            uint4 z = {0u, 0u, 0u, 0u};
            kf0.q2[1] = z; kf1.q2[1] = z;
        }

        v8f sc0 = {}, sc1 = {};
        sc0 = __builtin_amdgcn_wmma_f32_16x16x32_bf16(false, qa.v, false, kf0.v,
                                                      (short)0, sc0, false, false);
        sc1 = __builtin_amdgcn_wmma_f32_16x16x32_bf16(false, qa.v, false, kf1.v,
                                                      (short)0, sc1, false, false);

        // ALiBi + online softmax over 32 keys
        const int key0 = kt * 32 + r;
        float p0[8], p1[8];
        #pragma unroll
        for (int i = 0; i < 8; ++i) {
            int qi    = q0 + hlf * 8 + i;
            float sv0 = sc0[i] - slope * fabsf((float)(qi - key0));
            float sv1 = sc1[i] - slope * fabsf((float)(qi - key0 - 16));
            float mx  = fmaxf(sv0, sv1);
            #pragma unroll
            for (int m = 8; m; m >>= 1) mx = fmaxf(mx, __shfl_xor(mx, m, 32));
            float nm   = fmaxf(rm[i], mx);
            float corr = __expf(rm[i] - nm);
            p0[i] = __expf(sv0 - nm);
            p1[i] = __expf(sv1 - nm);
            float sum = p0[i] + p1[i];
            #pragma unroll
            for (int m = 8; m; m >>= 1) sum += __shfl_xor(sum, m, 32);
            rl[i]   = rl[i] * corr + sum;
            rm[i]   = nm;
            oacc[i] *= corr;
        }

        // stage probs (C-layout -> A-layout bounce through LDS)
        #pragma unroll
        for (int i = 0; i < 8; ++i) {
            ps[wave][hlf * 8 + i][r]      = f2bf(p0[i]);
            ps[wave][hlf * 8 + i][16 + r] = f2bf(p1[i]);
        }
        __syncthreads();

        // P (16x32) @ V (32x16): full K=32 WMMA, no padding
        BF16x16 pa, vf;
        pa.q2[0] = *(const uint4*)&ps[wave][r][hlf << 3];
        pa.q2[1] = *(const uint4*)&ps[wave][r][(hlf << 3) + 16];
        vf.q2[0] = *(const uint4*)&vsT[r][hlf << 3];
        vf.q2[1] = *(const uint4*)&vsT[r][(hlf << 3) + 16];

        oacc = __builtin_amdgcn_wmma_f32_16x16x32_bf16(false, pa.v, false, vf.v,
                                                       (short)0, oacc, false, false);
    }

    // normalize and scatter straight into [B,S,D] (undoes head transpose)
    #pragma unroll
    for (int i = 0; i < 8; ++i) {
        int qi = q0 + hlf * 8 + i;
        out[((size_t)bb * 512 + qi) * 256 + h * 16 + r] = oacc[i] / rl[i];
    }
}

// ---------------------------------------------------------------------------
// Deterministic split-K reduction + bias + GELU for the latent projection.
// ---------------------------------------------------------------------------
__global__ __launch_bounds__(256)
void reduce_bias_gelu_kernel(const float* __restrict__ parts, const float* __restrict__ bias,
                             float* __restrict__ out, int mn, int nsplit, int N)
{
    int t = blockIdx.x * blockDim.x + threadIdx.x;
    if (t >= mn) return;
    float s = 0.f;
    for (int z = 0; z < nsplit; ++z) s += parts[(size_t)z * mn + t];
    out[t] = gelu_f(s + bias[t % N]);
}

// ---------------------------------------------------------------------------
// Host launcher
// ---------------------------------------------------------------------------
extern "C" void kernel_launch(void* const* d_in, const int* in_sizes, int n_in,
                              void* d_out, int out_size, void* d_ws, size_t ws_size,
                              hipStream_t stream)
{
    const float* x        = (const float*)d_in[0];
    const float* pe_ln1_g = (const float*)d_in[1];
    const float* pe_ln1_b = (const float*)d_in[2];
    const float* pe_w     = (const float*)d_in[3];
    const float* pe_b     = (const float*)d_in[4];
    const float* pe_ln2_g = (const float*)d_in[5];
    const float* pe_ln2_b = (const float*)d_in[6];
    const float* Wq       = (const float*)d_in[7];
    const float* bq       = (const float*)d_in[8];
    const float* Wk       = (const float*)d_in[9];
    const float* bk       = (const float*)d_in[10];
    const float* Wv       = (const float*)d_in[11];
    const float* bv       = (const float*)d_in[12];
    const float* Wo       = (const float*)d_in[13];
    const float* bo       = (const float*)d_in[14];
    const float* W1       = (const float*)d_in[15];
    const float* b1w      = (const float*)d_in[16];
    const float* W2       = (const float*)d_in[17];
    const float* b2w      = (const float*)d_in[18];
    const float* ln1_g    = (const float*)d_in[19];
    const float* ln1_b    = (const float*)d_in[20];
    const float* ln2_g    = (const float*)d_in[21];
    const float* ln2_b    = (const float*)d_in[22];
    const float* latent_w = (const float*)d_in[23];
    const float* latent_b = (const float*)d_in[24];
    const float* final_w  = (const float*)d_in[25];
    const float* final_b  = (const float*)d_in[26];

    float* ws = (float*)d_ws;
    const size_t NB = (size_t)BB * SS * DDIM;     // 4,194,304 floats
    float* hbuf  = ws;                // residual stream [B*S, D]
    float* buf1  = ws + 1 * NB;       // attn_out
    float* buf2  = ws + 2 * NB;       // pre-LN residual sums
    float* qh    = ws + 3 * NB;       // [B,H,S,HD]
    float* kh    = ws + 4 * NB;
    float* vh    = ws + 5 * NB;
    float* ffmid = ws + 3 * NB;       // reuses qh+kh region after attention (8M floats)
    float* pln   = ws + 6 * NB;                          // [16384, 64]
    float* parts = ws + 6 * NB + (size_t)ROWS * PDIM;    // [64, 32, 256]
    float* zbuf  = parts + (size_t)64 * BB * LATD;       // [32, 256]

    const dim3 gTok(DDIM / 64, ROWS / 64, 1);     // (4, 256)
    const dim3 gFF (FFD  / 64, ROWS / 64, 1);     // (8, 256)

    // ---- patch embed ----
    patch_ln_kernel<<<ROWS / 8, 256, 0, stream>>>(x, pe_ln1_g, pe_ln1_b, pln);
    gemm64_bf16_wmma<true, false, false, 0><<<gTok, 128, 0, stream>>>(
        pln, pe_w, pe_b, nullptr, buf1, ROWS, DDIM, PDIM);
    layernorm_kernel<<<ROWS / 8, 256, 0, stream>>>(buf1, pe_ln2_g, pe_ln2_b, hbuf, ROWS, DDIM);

    // ---- transformer layers ----
    for (int l = 0; l < LLAY; ++l) {
        const size_t wo  = (size_t)l * DDIM * DDIM;
        const size_t bo_ = (size_t)l * DDIM;
        // Q/K with fused RoPE + [B,H,S,HD] scatter; V with fused scatter
        gemm64_bf16_wmma<true, false, false, 1><<<gTok, 128, 0, stream>>>(
            hbuf, Wq + wo, bq + bo_, nullptr, qh, ROWS, DDIM, DDIM);
        gemm64_bf16_wmma<true, false, false, 1><<<gTok, 128, 0, stream>>>(
            hbuf, Wk + wo, bk + bo_, nullptr, kh, ROWS, DDIM, DDIM);
        gemm64_bf16_wmma<true, false, false, 2><<<gTok, 128, 0, stream>>>(
            hbuf, Wv + wo, bv + bo_, nullptr, vh, ROWS, DDIM, DDIM);

        flash_attn_kernel<<<BB * HH * (SS / 64), 128, 0, stream>>>(qh, kh, vh, buf1);

        gemm64_bf16_wmma<true, false, true, 0><<<gTok, 128, 0, stream>>>(
            buf1, Wo + wo, bo + bo_, hbuf, buf2, ROWS, DDIM, DDIM);
        layernorm_kernel<<<ROWS / 8, 256, 0, stream>>>(buf2, ln1_g + bo_, ln1_b + bo_,
                                                       hbuf, ROWS, DDIM);

        gemm64_bf16_wmma<true, true, false, 0><<<gFF, 128, 0, stream>>>(
            hbuf, W1 + (size_t)l * DDIM * FFD, b1w + (size_t)l * FFD, nullptr,
            ffmid, ROWS, FFD, DDIM);
        gemm64_bf16_wmma<true, false, true, 0><<<gTok, 128, 0, stream>>>(
            ffmid, W2 + (size_t)l * FFD * DDIM, b2w + bo_, hbuf, buf2, ROWS, DDIM, FFD);
        layernorm_kernel<<<ROWS / 8, 256, 0, stream>>>(buf2, ln2_g + bo_, ln2_b + bo_,
                                                       hbuf, ROWS, DDIM);
    }

    // ---- latent projection: [32, 131072] @ [131072, 256], deterministic split-K ----
    gemm32_bf16_wmma<false, true><<<dim3(LATD / 32, 1, 64), 128, 0, stream>>>(
        hbuf, latent_w, nullptr, parts, BB, LATD, SS * DDIM, 2048);
    reduce_bias_gelu_kernel<<<(BB * LATD) / 256, 256, 0, stream>>>(
        parts, latent_b, zbuf, BB * LATD, 64, LATD);

    // ---- final head -> d_out (f32) ----
    gemm32_bf16_wmma<true, false><<<dim3(LATD / 32, 1, 1), 128, 0, stream>>>(
        zbuf, final_w, final_b, (float*)d_out, BB, LATD, LATD, 0);
}
